// BDHLinearAttention_27462020890945
// MI455X (gfx1250) — matrix operand order repaired and checked by reading, not compiled, not run on hardware
//
#include <hip/hip_runtime.h>
#include <hip/hip_bf16.h>

typedef __bf16 bf16_t;
typedef __attribute__((ext_vector_type(16))) __bf16 v16bf;
typedef __attribute__((ext_vector_type(8)))  __bf16 v8bf;
typedef __attribute__((ext_vector_type(8)))  float  v8f;

#define B_   8
#define N_   1024
#define D_   768
#define H_   8
#define S_   3072
#define HD_  384
#define MROWS (B_*N_)      // 8192
#define NCOL3S (3*S_)      // 9216

static __device__ __forceinline__ bf16_t f2bf(float f) {
  __hip_bfloat16 h = __float2bfloat16(f);
  return *reinterpret_cast<bf16_t*>(&h);
}
static __device__ __forceinline__ float bf2f(bf16_t v) {
  __hip_bfloat16 h = *reinterpret_cast<__hip_bfloat16*>(&v);
  return __bfloat162float(h);
}
static __device__ __forceinline__ float sigmoidf_(float x) {
  return 1.0f / (1.0f + expf(-x));
}

// ---------------------------------------------------------------------------
// Shared bf16 WMMA GEMM mainloop (double-buffered LDS, 1 barrier per K-step,
// short staging live-ranges to avoid VGPR spills).
//  C[128x128] tile = A[rowBase:+128, 0:K] (row-major, ld=lda, contiguous in K)
//                  @ Bt[colBase:+128, 0:K]^T (Bt row-major over [col][k])
//  256 threads = 8 waves in a 4(M) x 2(N) grid; each wave owns 32x64 = 2x4
//  16x16 tiles -> 8 v_wmma_f32_16x16x32_bf16 per 32-wide K step.
// ---------------------------------------------------------------------------
__device__ __forceinline__ void gemm_tile_bf16(
    const bf16_t* __restrict__ A, int lda,
    const bf16_t* __restrict__ Bt, int ldb,
    int K, int rowBase, int colBase,
    v8f (&acc)[2][4])
{
  // 80B row stride: 16B aligned for ds_load_b128, 20-bank stride (conflict-free)
  __shared__ __align__(16) bf16_t As[2][128][40];
  __shared__ __align__(16) bf16_t Bs[2][128][40];
  const int tid  = threadIdx.x;
  const int lane = tid & 31;
  const int wave = tid >> 5;
  const int waveM = wave >> 1;   // 0..3
  const int waveN = wave & 1;    // 0..1

#pragma unroll
  for (int tm = 0; tm < 2; ++tm)
#pragma unroll
    for (int tn = 0; tn < 4; ++tn)
#pragma unroll
      for (int i = 0; i < 8; ++i)
        acc[tm][tn][i] = 0.0f;

  const int akb = (lane < 16) ? 0 : 8;    // A K-run base (ISA 16-bit A layout)
  const int bkb = (lane < 16) ? 0 : 8;    // B K-run base

  // staging coordinates: 512 16B chunks per 128x32 tile, 2 per thread
  int cr[2], cc[2];
#pragma unroll
  for (int i = 0; i < 2; ++i) {
    int chunk = tid + i * 256;
    cr[i] = chunk >> 2;
    cc[i] = (chunk & 3) * 8;
  }

  // preload first K-slice into buffer 0
#pragma unroll
  for (int i = 0; i < 2; ++i) {
    *(v8bf*)&As[0][cr[i]][cc[i]] =
        *(const v8bf*)(A  + (size_t)(rowBase + cr[i]) * lda + cc[i]);
    *(v8bf*)&Bs[0][cr[i]][cc[i]] =
        *(const v8bf*)(Bt + (size_t)(colBase + cr[i]) * ldb + cc[i]);
  }
  __syncthreads();

  int p = 0;
  for (int k0 = 0; k0 < K; k0 += 32) {
    // A fragments for this wave's two row-tiles (16 VGPRs live)
    v16bf afrag[2];
#pragma unroll
    for (int tm = 0; tm < 2; ++tm) {
      int r = waveM * 32 + tm * 16 + (lane & 15);
      union { v16bf v; v8bf h[2]; } u;
      u.h[0] = *(const v8bf*)&As[p][r][akb];        // K = akb..akb+7
      u.h[1] = *(const v8bf*)&As[p][r][akb + 16];   // K = akb+16..akb+23
      afrag[tm] = u.v;
    }
    // one B fragment at a time, consumed immediately by both row-tiles
#pragma unroll
    for (int tn = 0; tn < 4; ++tn) {
      int r = waveN * 64 + tn * 16 + (lane & 15);
      union { v16bf v; v8bf h[2]; } u;
      u.h[0] = *(const v8bf*)&Bs[p][r][bkb];        // K = bkb..bkb+7
      u.h[1] = *(const v8bf*)&Bs[p][r][bkb + 16];   // K = bkb+16..bkb+23
      acc[0][tn] = __builtin_amdgcn_wmma_f32_16x16x32_bf16(
          false, afrag[0], false, u.v, (short)0, acc[0][tn], false, false);
      acc[1][tn] = __builtin_amdgcn_wmma_f32_16x16x32_bf16(
          false, afrag[1], false, u.v, (short)0, acc[1][tn], false, false);
    }

    // stage next K-slice into the other buffer (transient load->store pairs;
    // scheduler may hoist the global loads as pressure allows).  Single
    // barrier covers "stores visible before next reads" and "reads of p done
    // before p is overwritten next step" (__syncthreads => s_wait_dscnt 0).
    if ((k0 + 32) < K) {
#pragma unroll
      for (int i = 0; i < 2; ++i) {
        v8bf ta = *(const v8bf*)(A  + (size_t)(rowBase + cr[i]) * lda + (k0 + 32 + cc[i]));
        *(v8bf*)&As[p ^ 1][cr[i]][cc[i]] = ta;
        v8bf tb = *(const v8bf*)(Bt + (size_t)(colBase + cr[i]) * ldb + (k0 + 32 + cc[i]));
        *(v8bf*)&Bs[p ^ 1][cr[i]][cc[i]] = tb;
      }
      __syncthreads();
      p ^= 1;
    }
  }
}

// ---------------------------------------------------------------------------
// 1) LayerNorm -> xn (bf16) ; residue = sigmoid(xn @ w_res + b_res)
// ---------------------------------------------------------------------------
__global__ void ln_kernel(const float* __restrict__ x,
                          const float* __restrict__ ln_g,
                          const float* __restrict__ ln_b,
                          const float* __restrict__ w_res,
                          const float* __restrict__ b_res,
                          bf16_t* __restrict__ xn_bf,
                          float* __restrict__ residue)
{
  __shared__ float sred[256];
  const int tid = threadIdx.x;
  const int row = blockIdx.x;
  const float* xr = x + (size_t)row * D_;

  float xs[3];
  float s = 0.f, ss = 0.f;
#pragma unroll
  for (int i = 0; i < 3; ++i) {
    xs[i] = xr[tid + i * 256];
    s  += xs[i];
    ss += xs[i] * xs[i];
  }
  sred[tid] = s; __syncthreads();
  for (int off = 128; off > 0; off >>= 1) {
    if (tid < off) sred[tid] += sred[tid + off];
    __syncthreads();
  }
  float mean = sred[0] * (1.0f / D_);
  __syncthreads();
  sred[tid] = ss; __syncthreads();
  for (int off = 128; off > 0; off >>= 1) {
    if (tid < off) sred[tid] += sred[tid + off];
    __syncthreads();
  }
  float var  = sred[0] * (1.0f / D_) - mean * mean;   // biased variance
  __syncthreads();
  float rstd = rsqrtf(var + 1e-5f);

  float rdot = 0.f;
#pragma unroll
  for (int i = 0; i < 3; ++i) {
    int c = tid + i * 256;
    float xn = (xs[i] - mean) * rstd * ln_g[c] + ln_b[c];
    xn_bf[(size_t)row * D_ + c] = f2bf(xn);
    rdot += xn * w_res[c];
  }
  sred[tid] = rdot; __syncthreads();
  for (int off = 128; off > 0; off >>= 1) {
    if (tid < off) sred[tid] += sred[tid + off];
    __syncthreads();
  }
  if (tid == 0) residue[row] = sigmoidf_(sred[0] + b_res[0]);
}

// ---------------------------------------------------------------------------
// 2) Transpose fp32 [R x C] -> bf16 [C x R] (32x32 LDS tiles, coalesced)
// ---------------------------------------------------------------------------
__global__ void transpose_kernel(const float* __restrict__ in,
                                 bf16_t* __restrict__ out, int R, int C)
{
  __shared__ float tile[32][33];
  const int tc = blockIdx.x * 32;
  const int tr = blockIdx.y * 32;
  const int tx = threadIdx.x & 31;
  const int ty = threadIdx.x >> 5;   // 0..7
#pragma unroll
  for (int i = 0; i < 4; ++i) {
    int r = ty + i * 8;
    tile[r][tx] = in[(size_t)(tr + r) * C + tc + tx];
  }
  __syncthreads();
#pragma unroll
  for (int i = 0; i < 4; ++i) {
    int r = ty + i * 8;
    out[(size_t)(tc + r) * R + tr + tx] = f2bf(tile[tx][r]);
  }
}

// ---------------------------------------------------------------------------
// 3) Token mean of xn per (b, channel)  ->  xmean [B, D]
// ---------------------------------------------------------------------------
__global__ void xmean_kernel(const bf16_t* __restrict__ xn, float* __restrict__ xmean)
{
  int idx = blockIdx.x * 256 + threadIdx.x;    // 6144 total
  int b = idx / D_, c = idx % D_;
  const bf16_t* base = xn + (size_t)b * N_ * D_ + c;
  float s = 0.f;
  for (int n = 0; n < N_; ++n) s += bf2f(base[(size_t)n * D_]);
  xmean[idx] = s * (1.0f / N_);
}

// ---------------------------------------------------------------------------
// 4) read/write gates + wmean (one small block)
// ---------------------------------------------------------------------------
__global__ void gates_kernel(const float* __restrict__ xmean,
                             const float* __restrict__ w_rg, const float* __restrict__ b_rg,
                             const float* __restrict__ w_wg, const float* __restrict__ b_wg,
                             float* __restrict__ readg, float* __restrict__ wmean)
{
  __shared__ float wsh[64];
  const int t = threadIdx.x;   // 64 threads
  int b = t >> 3, h = t & 7;
  float r = 0.f, w = 0.f;
  for (int d = 0; d < D_; ++d) {
    float xm = xmean[b * D_ + d];
    r += xm * w_rg[d * H_ + h];
    w += xm * w_wg[d * H_ + h];
  }
  readg[t] = sigmoidf_(r + b_rg[h]);
  wsh[t]   = sigmoidf_(w + b_wg[h]);
  __syncthreads();
  if (t < H_) {
    float s = 0.f;
    for (int bb = 0; bb < B_; ++bb) s += wsh[bb * H_ + t];
    wmean[t] = s * (1.0f / B_);
  }
}

// ---------------------------------------------------------------------------
// 5) qkv GEMM: [8192 x 768] @ [768 x 9216] + fused feature-map epilogue.
//    q -> qbuf row-major bf16 [b,n,s]; k,v -> transposed bf16 [b,h,d,n]
//    (per-lane packed v8bf stores: g-axis == n-axis is contiguous).
// ---------------------------------------------------------------------------
__global__ void qkv_kernel(const bf16_t* __restrict__ xn, const bf16_t* __restrict__ w_inT,
                           const float* __restrict__ b_in,
                           bf16_t* __restrict__ qbuf, bf16_t* __restrict__ kT,
                           bf16_t* __restrict__ vT)
{
  v8f acc[2][4];
  const int rowBase = blockIdx.y * 128;
  const int colBase = blockIdx.x * 128;
  gemm_tile_bf16(xn, D_, w_inT, D_, D_, rowBase, colBase, acc);

  const int lane = threadIdx.x & 31;
  const int wave = threadIdx.x >> 5;
  const int waveM = wave >> 1, waveN = wave & 1;
  const int sec = colBase / S_;            // uniform per block (S_%128==0)

#pragma unroll
  for (int tm = 0; tm < 2; ++tm) {
#pragma unroll
    for (int tn = 0; tn < 4; ++tn) {
      int col = colBase + waveN * 64 + tn * 16 + (lane & 15);
      int within = col - sec * S_;
      int r0 = rowBase + waveM * 32 + tm * 16 + ((lane & 16) ? 8 : 0);
      float bias = b_in[col];
      if (sec == 0) {
#pragma unroll
        for (int g = 0; g < 8; ++g) {
          float val = acc[tm][tn][g] + bias;
          float t = fmaxf(val, 0.f) + 1e-6f;
          qbuf[(size_t)(r0 + g) * S_ + within] = f2bf(t * sqrtf(t));
        }
      } else {
        int h = within / HD_, d = within % HD_;
        int b = r0 >> 10, n0 = r0 & (N_ - 1);
        v8bf pk;
#pragma unroll
        for (int g = 0; g < 8; ++g) {
          float val = acc[tm][tn][g] + bias;
          if (sec == 1) { float t = fmaxf(val, 0.f) + 1e-6f; val = t * sqrtf(t); }
          pk[g] = f2bf(val);
        }
        bf16_t* dst = (sec == 1 ? kT : vT)
                    + ((size_t)(b * H_ + h) * HD_ + d) * N_ + n0;
        *(v8bf*)dst = pk;
      }
    }
  }
}

// ---------------------------------------------------------------------------
// 6) ksum[b,h,d] = sum_n kT[b,h,d,n]  (one wave per row)
// ---------------------------------------------------------------------------
__global__ void ksum_kernel(const bf16_t* __restrict__ kT, float* __restrict__ ksum)
{
  const int wave = threadIdx.x >> 5;
  const int lane = threadIdx.x & 31;
  const int row = blockIdx.x * 8 + wave;   // 24576 rows
  const bf16_t* src = kT + (size_t)row * N_;
  float s = 0.f;
  for (int i = lane; i < N_; i += 32) s += bf2f(src[i]);
  for (int m = 16; m > 0; m >>= 1) s += __shfl_xor(s, m, 32);
  if (lane == 0) ksum[row] = s;
}

// ---------------------------------------------------------------------------
// 7) state GEMM per (b,h): state[d,e] = (sum_n kT[d,n] vT[e,n]) / (ksum[d]+eps)
// ---------------------------------------------------------------------------
__global__ void state_kernel(const bf16_t* __restrict__ kT, const bf16_t* __restrict__ vT,
                             const float* __restrict__ ksum, float* __restrict__ state)
{
  const int bh = blockIdx.z;
  const bf16_t* A  = kT + (size_t)bh * HD_ * N_;
  const bf16_t* Bt = vT + (size_t)bh * HD_ * N_;
  v8f acc[2][4];
  const int rowBase = blockIdx.y * 128;
  const int colBase = blockIdx.x * 128;
  gemm_tile_bf16(A, N_, Bt, N_, N_, rowBase, colBase, acc);

  const int lane = threadIdx.x & 31;
  const int wave = threadIdx.x >> 5;
  const int waveM = wave >> 1, waveN = wave & 1;
#pragma unroll
  for (int tm = 0; tm < 2; ++tm) {
#pragma unroll
    for (int tn = 0; tn < 4; ++tn) {
      int e  = colBase + waveN * 64 + tn * 16 + (lane & 15);
      int d0 = rowBase + waveM * 32 + tm * 16 + ((lane & 16) ? 8 : 0);
#pragma unroll
      for (int g = 0; g < 8; ++g) {
        int d = d0 + g;
        float kd = ksum[bh * HD_ + d] + 1e-6f;
        state[((size_t)bh * HD_ + d) * HD_ + e] = acc[tm][tn][g] / kd;
      }
    }
  }
}

// ---------------------------------------------------------------------------
// 8) EMA memory update -> mem_newT bf16 [h, e, d]  (transposed for y GEMM)
// ---------------------------------------------------------------------------
__global__ void mem_kernel(const float* __restrict__ state, const float* __restrict__ memory,
                           const float* __restrict__ wmean, bf16_t* __restrict__ memT)
{
  int idx = blockIdx.x * 256 + threadIdx.x;      // 8*384*384 = 1179648 exactly
  int h = idx / (HD_ * HD_);
  int r = idx % (HD_ * HD_);
  int d = r / HD_, e = r % HD_;
  float s = 0.f;
  for (int b = 0; b < B_; ++b)
    s += state[((size_t)(b * H_ + h) * HD_ + d) * HD_ + e];
  float val = 0.95f * memory[idx] + 0.05f * wmean[h] * (s * (1.0f / B_));
  memT[((size_t)h * HD_ + e) * HD_ + d] = f2bf(val);
}

// ---------------------------------------------------------------------------
// 9) y GEMM per (b,h): y[n,e] = read[b,h] * sum_d q[n,d] * mem_new[d,e]
// ---------------------------------------------------------------------------
__global__ void y_kernel(const bf16_t* __restrict__ qbuf, const bf16_t* __restrict__ memT,
                         const float* __restrict__ readg, bf16_t* __restrict__ ybuf)
{
  const int bh = blockIdx.z;
  const int b = bh >> 3, h = bh & 7;
  const bf16_t* A  = qbuf + (size_t)b * N_ * S_ + h * HD_;
  const bf16_t* Bt = memT + (size_t)h * HD_ * HD_;
  v8f acc[2][4];
  const int rowBase = blockIdx.y * 128;
  const int colBase = blockIdx.x * 128;
  gemm_tile_bf16(A, S_, Bt, HD_, HD_, rowBase, colBase, acc);

  const float rg = readg[bh];
  const int lane = threadIdx.x & 31;
  const int wave = threadIdx.x >> 5;
  const int waveM = wave >> 1, waveN = wave & 1;
#pragma unroll
  for (int tm = 0; tm < 2; ++tm) {
#pragma unroll
    for (int tn = 0; tn < 4; ++tn) {
      int e  = colBase + waveN * 64 + tn * 16 + (lane & 15);
      int n0 = rowBase + waveM * 32 + tm * 16 + ((lane & 16) ? 8 : 0);
#pragma unroll
      for (int g = 0; g < 8; ++g)
        ybuf[(size_t)(b * N_ + n0 + g) * S_ + h * HD_ + e] = f2bf(acc[tm][tn][g] * rg);
    }
  }
}

// ---------------------------------------------------------------------------
// 10) out GEMM: out = residue*(y @ w_out + b_out) + (1-residue)*x
// ---------------------------------------------------------------------------
__global__ void out_kernel(const bf16_t* __restrict__ ybuf, const bf16_t* __restrict__ w_outT,
                           const float* __restrict__ b_out, const float* __restrict__ residue,
                           const float* __restrict__ x, float* __restrict__ out)
{
  v8f acc[2][4];
  const int rowBase = blockIdx.y * 128;
  const int colBase = blockIdx.x * 128;
  gemm_tile_bf16(ybuf, S_, w_outT, S_, S_, rowBase, colBase, acc);

  const int lane = threadIdx.x & 31;
  const int wave = threadIdx.x >> 5;
  const int waveM = wave >> 1, waveN = wave & 1;
#pragma unroll
  for (int tm = 0; tm < 2; ++tm) {
#pragma unroll
    for (int tn = 0; tn < 4; ++tn) {
      int col = colBase + waveN * 64 + tn * 16 + (lane & 15);
      int r0  = rowBase + waveM * 32 + tm * 16 + ((lane & 16) ? 8 : 0);
      float bias = b_out[col];
#pragma unroll
      for (int g = 0; g < 8; ++g) {
        int grow = r0 + g;
        float res = residue[grow];
        float val = res * (acc[tm][tn][g] + bias)
                  + (1.0f - res) * x[(size_t)grow * D_ + col];
        out[(size_t)grow * D_ + col] = val;
      }
    }
  }
}

// ---------------------------------------------------------------------------
extern "C" void kernel_launch(void* const* d_in, const int* in_sizes, int n_in,
                              void* d_out, int out_size, void* d_ws, size_t ws_size,
                              hipStream_t stream) {
  const float* x      = (const float*)d_in[0];
  const float* memory = (const float*)d_in[1];
  const float* ln_g   = (const float*)d_in[2];
  const float* ln_b   = (const float*)d_in[3];
  const float* w_in   = (const float*)d_in[4];
  const float* b_in   = (const float*)d_in[5];
  const float* w_out  = (const float*)d_in[6];
  const float* b_out  = (const float*)d_in[7];
  const float* w_rg   = (const float*)d_in[8];
  const float* b_rg   = (const float*)d_in[9];
  const float* w_wg   = (const float*)d_in[10];
  const float* b_wg   = (const float*)d_in[11];
  const float* w_res  = (const float*)d_in[12];
  const float* b_res  = (const float*)d_in[13];
  float* out = (float*)d_out;

  char* ws = (char*)d_ws;
  size_t off = 0;
  auto carve = [&](size_t bytes) -> char* {
    char* p = ws + off;
    off += (bytes + 255) & ~(size_t)255;
    return p;
  };

  bf16_t* xn_bf  = (bf16_t*)carve((size_t)MROWS * D_ * 2);          // 12.6 MB
  bf16_t* w_inT  = (bf16_t*)carve((size_t)NCOL3S * D_ * 2);         // 14.2 MB
  bf16_t* w_outT = (bf16_t*)carve((size_t)D_ * S_ * 2);             //  4.7 MB
  bf16_t* qbuf   = (bf16_t*)carve((size_t)MROWS * S_ * 2);          // 50.3 MB
  bf16_t* kT     = (bf16_t*)carve((size_t)B_ * H_ * HD_ * N_ * 2);  // 50.3 MB
  bf16_t* vT     = (bf16_t*)carve((size_t)B_ * H_ * HD_ * N_ * 2);  // 50.3 MB
  bf16_t* ybuf   = (bf16_t*)carve((size_t)MROWS * S_ * 2);          // 50.3 MB
  float*  statef = (float*) carve((size_t)B_ * H_ * HD_ * HD_ * 4); // 37.7 MB
  bf16_t* memT   = (bf16_t*)carve((size_t)H_ * HD_ * HD_ * 2);      //  2.4 MB
  float*  ksum   = (float*) carve((size_t)B_ * H_ * HD_ * 4);
  float*  xmean  = (float*) carve((size_t)B_ * D_ * 4);
  float*  residue= (float*) carve((size_t)MROWS * 4);
  float*  readg  = (float*) carve((size_t)B_ * H_ * 4);
  float*  wmean  = (float*) carve((size_t)H_ * 4);
  (void)ws_size; (void)in_sizes; (void)n_in; (void)out_size;

  // 1) LayerNorm + residue gate
  ln_kernel<<<MROWS, 256, 0, stream>>>(x, ln_g, ln_b, w_res, b_res, xn_bf, residue);
  // 2) Weight packs (transpose + bf16)
  transpose_kernel<<<dim3(NCOL3S / 32, D_ / 32), 256, 0, stream>>>(w_in, w_inT, D_, NCOL3S);
  transpose_kernel<<<dim3(D_ / 32, S_ / 32), 256, 0, stream>>>(w_out, w_outT, S_, D_);
  // 3) Token means, 4) gates
  xmean_kernel<<<(B_ * D_) / 256, 256, 0, stream>>>(xn_bf, xmean);
  gates_kernel<<<1, 64, 0, stream>>>(xmean, w_rg, b_rg, w_wg, b_wg, readg, wmean);
  // 5) qkv GEMM + feature-map epilogue
  qkv_kernel<<<dim3(NCOL3S / 128, MROWS / 128), 256, 0, stream>>>(
      xn_bf, w_inT, b_in, qbuf, kT, vT);
  // 6) k column sums
  ksum_kernel<<<(B_ * H_ * HD_) / 8, 256, 0, stream>>>(kT, ksum);
  // 7) per-(b,h) state GEMM with fused normalization
  state_kernel<<<dim3(HD_ / 128, HD_ / 128, B_ * H_), 256, 0, stream>>>(
      kT, vT, ksum, statef);
  // 8) EMA memory update (transposed bf16 out)
  mem_kernel<<<(H_ * HD_ * HD_) / 256, 256, 0, stream>>>(statef, memory, wmean, memT);
  // 9) per-(b,h) read GEMM
  y_kernel<<<dim3(HD_ / 128, N_ / 128, B_ * H_), 256, 0, stream>>>(
      qbuf, memT, readg, ybuf);
  // 10) out projection + residual blend
  out_kernel<<<dim3(D_ / 128, MROWS / 128), 256, 0, stream>>>(
      ybuf, w_outT, b_out, residue, x, out);
}
